// MultiPerspectiveLayer_41077067219604
// MI455X (gfx1250) — matrix operand order repaired and checked by reading, not compiled, not run on hardware
//
#include <hip/hip_runtime.h>
#include <hip/hip_bf16.h>

// Problem constants (from reference setup_inputs)
#define BQ   64
#define LQ   128
#define DQ   200
#define DH   100      // D/2
#define DP   224      // D padded to multiple of 32 for WMMA K-loop
#define MPQ  20
#define ATTQ 50
#define OUTC 83       // 4*MP + 3

typedef __bf16 v16bf __attribute__((ext_vector_type(16)));
typedef float  v8f   __attribute__((ext_vector_type(8)));
typedef unsigned int u32;
typedef u32 u32x4 __attribute__((ext_vector_type(4)));
typedef u32 u32x8 __attribute__((ext_vector_type(8)));

union H8 { uint4 u; __bf16 h[8]; };

static __device__ __forceinline__ float bf2f(__bf16 x) { return (float)x; }
static __device__ __forceinline__ __bf16 f2bf(float x) { return (__bf16)x; }

// ---------------------------------------------------------------------------
// TDM: issue a 2-D tensor DMA (bf16 rows x DP, stride DP) global -> LDS.
// D# group0 (128b) + group1 (256b) built in SGPRs; groups 2/3 omitted (2-D).
// Tracked by TENSORcnt.
// ---------------------------------------------------------------------------
static __device__ __forceinline__ void tdm_load_2d(const void* gaddr,
                                                   u32 lds_off, u32 rows)
{
    unsigned long long ga = (unsigned long long)(uintptr_t)gaddr;
    u32x4 g0;
    g0[0] = 1u;                                           // count=1 (valid), load
    g0[1] = lds_off;                                      // lds_addr (bytes)
    g0[2] = (u32)(ga & 0xFFFFFFFFu);                      // global_addr[31:0]
    g0[3] = (u32)((ga >> 32) & 0x01FFFFFFu) | (2u << 30); // addr[56:32] | type=2
    u32x8 g1;
    g1[0] = (1u << 16);          // workgroup_mask=0 | data_size=1 (2 bytes)
    g1[1] = ((u32)DP << 16);     // atomic_barrier_addr=0 | tensor_dim0.lo16
    g1[2] = (rows << 16);        // tensor_dim0.hi16=0 | tensor_dim1.lo16
    g1[3] = ((u32)DP << 16);     // tensor_dim1.hi16=0 | tile_dim0
    g1[4] = rows;                // tile_dim1 | tile_dim2=0
    g1[5] = (u32)DP;             // tensor_dim0_stride.lo32
    g1[6] = 0u;                  // stride0.hi16 | tensor_dim1_stride.lo16
    g1[7] = 0u;                  // tensor_dim1_stride.hi32
    asm volatile("tensor_load_to_lds %0, %1" :: "s"(g0), "s"(g1) : "memory");
}

// ---------------------------------------------------------------------------
// K0: convert reps_lt / reps_rt to zero-padded bf16 [B*L][DP]
// ---------------------------------------------------------------------------
__global__ __launch_bounds__(256) void convert_bf16_kernel(
    const float* __restrict__ lt, const float* __restrict__ rt,
    __bf16* __restrict__ ltb, __bf16* __restrict__ rtb)
{
    int i = blockIdx.x * blockDim.x + threadIdx.x;
    if (i >= BQ * LQ * DP) return;
    int d = i % DP;
    int row = i / DP;
    float lv = (d < DQ) ? lt[(size_t)row * DQ + d] : 0.0f;
    float rv = (d < DQ) ? rt[(size_t)row * DQ + d] : 0.0f;
    ltb[i] = f2bf(lv);
    rtb[i] = f2bf(rv);
}

// ---------------------------------------------------------------------------
// K1: row norms of reps_lt (for max-attentive relevancy)
// ---------------------------------------------------------------------------
__global__ __launch_bounds__(256) void norm_kernel(
    const float* __restrict__ lt, float* __restrict__ nlt)
{
    int i = blockIdx.x * blockDim.x + threadIdx.x;
    if (i >= BQ * LQ) return;
    const float* p = lt + (size_t)i * DQ;
    float s = 0.f;
    for (int d = 0; d < DQ; ++d) s += p[d] * p[d];
    nlt[i] = sqrtf(fmaxf(s, 1e-6f));
}

// ---------------------------------------------------------------------------
// K2: full match -> out cols [0 .. 20]
// ---------------------------------------------------------------------------
__global__ __launch_bounds__(128) void full_match_kernel(
    const float* __restrict__ lt, const float* __restrict__ fw,
    const float* __restrict__ bw, const float* __restrict__ wfull,
    float* __restrict__ out)
{
    int idx = blockIdx.x * blockDim.x + threadIdx.x;   // over B*L
    if (idx >= BQ * LQ) return;
    int b = idx / LQ;
    const float* ltr = lt + (size_t)idx * DQ;
    __builtin_prefetch(ltr, 0, 3);                     // global_prefetch_b8
    float acc[MPQ + 1];
#pragma unroll
    for (int m = 0; m <= MPQ; ++m) acc[m] = 0.f;
    for (int d = 0; d < DQ; ++d) {
        float h = (d < DH) ? fw[b * DH + d] : bw[b * DH + d - DH];
        float p = ltr[d] * h;
        acc[0] += p;
#pragma unroll
        for (int m = 0; m < MPQ; ++m) acc[m + 1] += p * wfull[m * DQ + d];
    }
    float* o = out + (size_t)idx * OUTC;
#pragma unroll
    for (int m = 0; m <= MPQ; ++m) o[m] = tanhf(acc[m]);
}

// ---------------------------------------------------------------------------
// K3: maxpool match via TDM + WMMA bf16 -> out cols [21 .. 40]
//   grid = (Ltiles=8, B=64), block = 256 (8 waves, wave w = K-tile w)
//   TDM stages rt block [128 x DP] and lt tile [16 x DP] into LDS once,
//   then loops over all 20 perspectives (7 wmma per wave per m).
//   S[l,k] = sum_d lt[b,l,d] * w[m,d] * rt[b,k,d]; out = tanh(max_k S)
// ---------------------------------------------------------------------------
__global__ __launch_bounds__(256) void maxpool_wmma_kernel(
    const __bf16* __restrict__ ltb, const __bf16* __restrict__ rtb,
    const float* __restrict__ w, float* __restrict__ out)
{
    extern __shared__ __align__(16) char smem_raw[];
    __bf16* Bsh = (__bf16*)smem_raw;                                // [LQ][DP]
    __bf16* Ash = (__bf16*)(smem_raw + (size_t)LQ * DP * 2);        // [16][DP]
    float (*smax)[16] =
        (float (*)[16])(smem_raw + (size_t)LQ * DP * 2 + 16 * DP * 2);

    const int ltile = blockIdx.x;           // 0..7
    const int b     = blockIdx.y;           // 0..63
    const int tid   = threadIdx.x;
    const int wave  = tid >> 5;             // 0..7 -> K-tile
    const int lane  = tid & 31;
    const int hi    = lane >> 4;            // lane group (0: 0-15, 1: 16-31)
    const int lrow  = lane & 15;

    // One wave issues both DMAs (TDM ignores EXEC; descriptors are uniform).
    if (wave == 0) {
        tdm_load_2d(rtb + (size_t)b * LQ * DP, 0u, (u32)LQ);
        tdm_load_2d(ltb + ((size_t)b * LQ + ltile * 16) * DP,
                    (u32)(LQ * DP * 2), 16u);
        __builtin_amdgcn_s_wait_tensorcnt(0);
    }
    __syncthreads();

    const __bf16* ARow = Ash + lrow * DP;
    const __bf16* BRow = Bsh + (wave * 16 + lrow) * DP;

    for (int m = 0; m < MPQ; ++m) {
        const float* wm = w + m * DQ;
        v8f c = {0.f, 0.f, 0.f, 0.f, 0.f, 0.f, 0.f, 0.f};

        for (int k0 = 0; k0 < DP; k0 += 32) {
            // --- A fragment (16x32 bf16): slots 0-7 -> K=k0+hi*8+j,
            //                              slots 8-15 -> K=k0+16+hi*8+j
            const int dA0 = k0 + hi * 8;
            const int dA1 = k0 + 16 + hi * 8;
            H8 a0, a1;
            a0.u = *reinterpret_cast<const uint4*>(ARow + dA0);
            a1.u = *reinterpret_cast<const uint4*>(ARow + dA1);
            v16bf A;
#pragma unroll
            for (int j = 0; j < 8; ++j) {
                int d0 = dA0 + j, d1 = dA1 + j;
                float s0 = (d0 < DQ) ? wm[d0] : 0.0f;
                float s1 = (d1 < DQ) ? wm[d1] : 0.0f;
                A[j]     = f2bf(bf2f(a0.h[j]) * s0);
                A[j + 8] = f2bf(bf2f(a1.h[j]) * s1);
            }
            // --- B fragment (32x16 bf16): slots 0-7 -> K=k0+hi*8+j, slots 8-15 -> K=k0+16+hi*8+j
            const int dB = k0 + hi * 8;
            H8 b0, b1;
            b0.u = *reinterpret_cast<const uint4*>(BRow + dB);
            b1.u = *reinterpret_cast<const uint4*>(BRow + dB + 16);
            v16bf Bf;
#pragma unroll
            for (int j = 0; j < 8; ++j) { Bf[j] = b0.h[j]; Bf[j + 8] = b1.h[j]; }

            c = __builtin_amdgcn_wmma_f32_16x16x32_bf16(
                    false, A, false, Bf, (short)0, c, false, false);
        }

        // Row-max over the 16 N columns held by this wave's tile.
        // c[v] @ lane: M = v + 8*hi, N = lane%16 -> reduce within 16-lane groups
        float rmax[8];
#pragma unroll
        for (int v = 0; v < 8; ++v) {
            float x = c[v];
            x = fmaxf(x, __shfl_xor(x, 8, 32));
            x = fmaxf(x, __shfl_xor(x, 4, 32));
            x = fmaxf(x, __shfl_xor(x, 2, 32));
            x = fmaxf(x, __shfl_xor(x, 1, 32));
            rmax[v] = x;
        }
        if (lrow == 0) {
#pragma unroll
            for (int v = 0; v < 8; ++v) smax[wave][v + hi * 8] = rmax[v];
        }
        __syncthreads();
        if (tid < 16) {
            float x = smax[0][tid];
#pragma unroll
            for (int wv = 1; wv < 8; ++wv) x = fmaxf(x, smax[wv][tid]);
            int l = ltile * 16 + tid;
            out[((size_t)b * LQ + l) * OUTC + 21 + m] = tanhf(x);
        }
        __syncthreads();
    }
}

// ---------------------------------------------------------------------------
// K4: attention projections e_rt = tanh(rt@w1)*diag, e_lt = tanh(lt@w2)
// ---------------------------------------------------------------------------
__global__ __launch_bounds__(256) void proj_kernel(
    const float* __restrict__ lt, const float* __restrict__ rt,
    const float* __restrict__ w1, const float* __restrict__ w2,
    const float* __restrict__ diag,
    float* __restrict__ e_rt, float* __restrict__ e_lt)
{
    int idx = blockIdx.x * blockDim.x + threadIdx.x;   // B*L*ATT
    if (idx >= BQ * LQ * ATTQ) return;
    int a = idx % ATTQ;
    int row = idx / ATTQ;
    const float* rtr = rt + (size_t)row * DQ;
    const float* ltr = lt + (size_t)row * DQ;
    float sr = 0.f, sl = 0.f;
    for (int d = 0; d < DQ; ++d) {
        sr += rtr[d] * w1[d * ATTQ + a];
        sl += ltr[d] * w2[d * ATTQ + a];
    }
    e_rt[idx] = tanhf(sr) * diag[a];
    e_lt[idx] = tanhf(sl);
}

// ---------------------------------------------------------------------------
// K5: attentive match (scores + softmax + attn*V + mp) -> out cols [41 .. 61]
//   one block per (b, r); 128 threads (one per left position)
// ---------------------------------------------------------------------------
__global__ __launch_bounds__(128) void attn_kernel(
    const float* __restrict__ lt, const float* __restrict__ rt,
    const float* __restrict__ e_rt, const float* __restrict__ e_lt,
    const float* __restrict__ w_att, float* __restrict__ out)
{
    const int r = blockIdx.x;
    const int b = blockIdx.y;
    const int t = threadIdx.x;

    __shared__ float p[LQ];
    __shared__ float att[DQ];
    __shared__ float red[LQ];

    const float* er = e_rt + ((size_t)b * LQ + r) * ATTQ;
    const float* el = e_lt + ((size_t)b * LQ + t) * ATTQ;
    float s = 0.f;
    for (int a = 0; a < ATTQ; ++a) s += er[a] * el[a];
    red[t] = s;
    __syncthreads();
    for (int off = 64; off >= 1; off >>= 1) {
        if (t < off) red[t] = fmaxf(red[t], red[t + off]);
        __syncthreads();
    }
    float mx = red[0];
    __syncthreads();
    float e = expf(s - mx);
    p[t] = e;
    red[t] = e;
    __syncthreads();
    for (int off = 64; off >= 1; off >>= 1) {
        if (t < off) red[t] += red[t + off];
        __syncthreads();
    }
    float inv = 1.0f / red[0];

    // att_lt[b,r,d] = sum_l softmax[l] * lt[b,l,d]
    for (int d = t; d < DQ; d += 128) {
        float a = 0.f;
        for (int l = 0; l < LQ; ++l) a += p[l] * lt[((size_t)b * LQ + l) * DQ + d];
        att[d] = a * inv;
    }
    __syncthreads();

    if (t <= MPQ) {
        const float* rr = rt + ((size_t)b * LQ + r) * DQ;
        float acc = 0.f;
        if (t == 0) {
            for (int d = 0; d < DQ; ++d) acc += att[d] * rr[d];
        } else {
            const float* wm = w_att + (t - 1) * DQ;
            for (int d = 0; d < DQ; ++d) acc += att[d] * wm[d] * rr[d];
        }
        out[((size_t)b * LQ + r) * OUTC + 41 + t] = tanhf(acc);
    }
}

// ---------------------------------------------------------------------------
// K6: max-attentive match -> out cols [62 .. 82]
//   one block per (b, r); argmax_l dot(rt_r, lt_l)/||lt_l|| then mp match
// ---------------------------------------------------------------------------
__global__ __launch_bounds__(128) void maxatt_kernel(
    const float* __restrict__ lt, const float* __restrict__ rt,
    const float* __restrict__ nlt, const float* __restrict__ w_ma,
    float* __restrict__ out)
{
    const int r = blockIdx.x;
    const int b = blockIdx.y;
    const int t = threadIdx.x;

    __shared__ float vals[LQ];
    __shared__ int   idxs[LQ];

    const float* rr = rt + ((size_t)b * LQ + r) * DQ;
    const float* lr = lt + ((size_t)b * LQ + t) * DQ;
    float s = 0.f;
    for (int d = 0; d < DQ; ++d) s += rr[d] * lr[d];
    vals[t] = s / nlt[b * LQ + t];
    idxs[t] = t;
    __syncthreads();
    for (int off = 64; off >= 1; off >>= 1) {
        if (t < off) {
            float v2 = vals[t + off];
            int   i2 = idxs[t + off];
            if (v2 > vals[t] || (v2 == vals[t] && i2 < idxs[t])) {
                vals[t] = v2; idxs[t] = i2;
            }
        }
        __syncthreads();
    }
    int pos = idxs[0];
    const float* la = lt + ((size_t)b * LQ + pos) * DQ;
    if (t <= MPQ) {
        float acc = 0.f;
        if (t == 0) {
            for (int d = 0; d < DQ; ++d) acc += rr[d] * la[d];
        } else {
            const float* wm = w_ma + (t - 1) * DQ;
            for (int d = 0; d < DQ; ++d) acc += rr[d] * wm[d] * la[d];
        }
        out[((size_t)b * LQ + r) * OUTC + 62 + t] = tanhf(acc);
    }
}

// ---------------------------------------------------------------------------
extern "C" void kernel_launch(void* const* d_in, const int* in_sizes, int n_in,
                              void* d_out, int out_size, void* d_ws, size_t ws_size,
                              hipStream_t stream) {
    (void)in_sizes; (void)n_in; (void)out_size; (void)ws_size;

    const float* reps_lt = (const float*)d_in[0];
    const float* reps_rt = (const float*)d_in[1];
    const float* fw_h_rt = (const float*)d_in[2];
    const float* bw_h_rt = (const float*)d_in[3];
    const float* w_full  = (const float*)d_in[4];
    const float* w_maxp  = (const float*)d_in[5];
    const float* w_att   = (const float*)d_in[6];
    const float* w_maxa  = (const float*)d_in[7];
    const float* attn_w1 = (const float*)d_in[8];
    const float* attn_w2 = (const float*)d_in[9];
    const float* diag_w  = (const float*)d_in[10];
    float* out = (float*)d_out;

    // Workspace carving (all chunks are 256B-aligned multiples)
    char* ws = (char*)d_ws;
    const size_t ltb_bytes = (size_t)BQ * LQ * DP * sizeof(__bf16);   // 3,670,016
    const size_t ert_bytes = (size_t)BQ * LQ * ATTQ * sizeof(float);  // 1,638,400
    __bf16* ltb  = (__bf16*)(ws);
    __bf16* rtb  = (__bf16*)(ws + ltb_bytes);
    float*  e_rt = (float*)(ws + 2 * ltb_bytes);
    float*  e_lt = (float*)(ws + 2 * ltb_bytes + ert_bytes);
    float*  nlt  = (float*)(ws + 2 * ltb_bytes + 2 * ert_bytes);

    // Stage 0: precompute bf16 copies + norms
    {
        int n = BQ * LQ * DP;
        convert_bf16_kernel<<<(n + 255) / 256, 256, 0, stream>>>(reps_lt, reps_rt, ltb, rtb);
    }
    norm_kernel<<<(BQ * LQ + 255) / 256, 256, 0, stream>>>(reps_lt, nlt);

    // Full match (cols 0..20)
    full_match_kernel<<<(BQ * LQ + 127) / 128, 128, 0, stream>>>(
        reps_lt, fw_h_rt, bw_h_rt, w_full, out);

    // Maxpool match via TDM + WMMA (cols 21..40)
    {
        size_t ldsBytes = (size_t)LQ * DP * 2 + 16 * DP * 2 + 8 * 16 * sizeof(float);
        maxpool_wmma_kernel<<<dim3(LQ / 16, BQ), 256, ldsBytes, stream>>>(
            ltb, rtb, w_maxp, out);
    }

    // Attentive match (cols 41..61)
    {
        int n = BQ * LQ * ATTQ;
        proj_kernel<<<(n + 255) / 256, 256, 0, stream>>>(
            reps_lt, reps_rt, attn_w1, attn_w2, diag_w, e_rt, e_lt);
    }
    attn_kernel<<<dim3(LQ, BQ), 128, 0, stream>>>(
        reps_lt, reps_rt, e_rt, e_lt, w_att, out);

    // Max-attentive match (cols 62..82)
    maxatt_kernel<<<dim3(LQ, BQ), 128, 0, stream>>>(
        reps_lt, reps_rt, nlt, w_maxa, out);
}
